// MultiHeadAttention_40080634806636
// MI455X (gfx1250) — matrix operand order repaired and checked by reading, not compiled, not run on hardware
//
#include <hip/hip_runtime.h>
#include <hip/hip_bf16.h>

// MI455X (gfx1250) fused MHA, round 3:
//  - WMMA bf16 GEMMs, B tiles staged block-wide in LDS via CDNA5 async
//    global->LDS copies (ASYNCcnt), double-buffered LDS
//  - raw split barrier (s_barrier_signal/-wait -1) instead of __syncthreads
//    in the GEMM loop: avoids the fence-induced s_wait_loadcnt 0 that was
//    draining the double-buffered A-fragment loads each K-step
//  - branchless wrap for next-iteration indices (kills s_cselect chains)
//  - flash attention with register double-buffered K/V fragments

typedef __attribute__((ext_vector_type(16))) __bf16 v16bf;
typedef __attribute__((ext_vector_type(8)))  __bf16 v8bf;
typedef __attribute__((ext_vector_type(8)))  float  v8f;

#define D_MODEL 1024
#define SEQ     2048
#define NHEADS  16
#define DK      64
#define BATCH   2

static __device__ __forceinline__ v8f wmma_bf16(v16bf a, v16bf b, v8f c) {
  return __builtin_amdgcn_wmma_f32_16x16x32_bf16(
      false, a, false, b, (short)0, c, false, false);
}

// A-fragment 16x32 bf16: per lane two contiguous 16B chunks (k=delta, delta+8)
static __device__ __forceinline__ v16bf load_frag_a(const __bf16* base) {
  v8bf lo = *(const v8bf*)(base);
  v8bf hi = *(const v8bf*)(base + 8);
  v16bf r;
#pragma unroll
  for (int i = 0; i < 8; ++i) { r[i] = lo[i]; r[i + 8] = hi[i]; }
  return r;
}

// B-fragment 32x16 bf16: per lane 32 contiguous bytes along K
static __device__ __forceinline__ v16bf load_frag_b(const __bf16* base) {
  return *(const v16bf*)(base);
}

static __device__ __forceinline__ v8f zero_v8f() {
  v8f z;
#pragma unroll
  for (int i = 0; i < 8; ++i) z[i] = 0.0f;
  return z;
}

// CDNA5 async global -> LDS copy (16B per lane), tracked by ASYNCcnt.
static __device__ __forceinline__ void async_b128_to_lds(const __bf16* gsrc,
                                                         unsigned lds_off) {
  asm volatile("global_load_async_to_lds_b128 %0, %1, off"
               :: "v"(lds_off), "v"(gsrc) : "memory");
}
static __device__ __forceinline__ void wait_asynccnt0() {
  asm volatile("s_wait_asynccnt 0x0" ::: "memory");
}

// Split workgroup barrier WITHOUT the acq/rel fence __syncthreads carries:
// no vmem-counter drain, so in-flight double-buffered loads stay in flight.
static __device__ __forceinline__ void block_bar() {
  asm volatile("s_barrier_signal -1\n\ts_barrier_wait -1" ::: "memory");
}

static __device__ __forceinline__ void wavebar() {
#if __has_builtin(__builtin_amdgcn_wave_barrier)
  __builtin_amdgcn_wave_barrier();
#else
  __syncthreads();
#endif
}

// ---------------------------------------------------------------- convert
__global__ void cvt_f32_bf16(const float* __restrict__ src,
                             __bf16* __restrict__ dst, int n) {
  int stride = gridDim.x * blockDim.x;
  for (int i = blockIdx.x * blockDim.x + threadIdx.x; i < n; i += stride)
    dst[i] = (__bf16)src[i];
}

// ---------------------------------------------------------------- GEMM
// C[m,n] = sum_k A[m,k] * W[n,k]   (torch Linear: x @ W^T)
// Block = 256 threads = 8 waves, covers 256 rows x 64 cols.
// Wave computes 32x64 (8 accumulators); B tile (64x32 bf16 = 4KB) staged
// block-wide in LDS by async copies (double-buffered); A frags double-buffered
// in registers.
// mode 0: bf16 [M,1024] | mode 1: bf16 Vt[b][h][d][s] | mode 2: f32 [M,1024]
__global__ __launch_bounds__(256) void gemm_wmma(
    const __bf16* __restrict__ A, const __bf16* __restrict__ W,
    void* __restrict__ out, int mode) {
  __shared__ __attribute__((aligned(128))) __bf16 Bs[2][64][32];  // 2 x 4KB

  const int lane = threadIdx.x & 31;
  const int wave = threadIdx.x >> 5;
  const int m0 = blockIdx.x * 256 + wave * 32;
  const int n0 = blockIdx.y * 64;
  const int hi = (lane >= 16);
  const int dA = hi ? 16 : 0;    // A-frag k delta (elements)
  const int dB2 = hi ? 32 : 0;   // B-frag k delta (bytes) within 64B LDS row

  // async-copy mapping: 256 threads x 16B = 4KB tile (64 rows x 64B)
  const int crow = threadIdx.x >> 2;           // 0..63  (n within tile)
  const int cchk = (threadIdx.x & 3) * 8;      // element offset within k-chunk
  const __bf16* gB = W + (size_t)(n0 + crow) * D_MODEL + cchk;
  const unsigned ldsBase = (unsigned)(uintptr_t)&Bs[0][0][0];
  const unsigned ldsRow = (unsigned)(crow * 64 + cchk * 2);

  // A rows for this wave (two 16-row blocks)
  const __bf16* aRow0 = A + (size_t)(m0 + (lane & 15)) * D_MODEL + dA;
  const __bf16* aRow1 = aRow0 + (size_t)16 * D_MODEL;

  // prologue: stage k0=0 B tile, load k0=0 A frags
  async_b128_to_lds(gB, ldsBase + ldsRow);
  v16bf a0 = load_frag_a(aRow0);
  v16bf a1 = load_frag_a(aRow1);
  wait_asynccnt0();
  block_bar();

  v8f acc[2][4];
#pragma unroll
  for (int rb = 0; rb < 2; ++rb)
#pragma unroll
    for (int t = 0; t < 4; ++t) acc[rb][t] = zero_v8f();

  for (int k0 = 0; k0 < D_MODEL; k0 += 32) {
    const int cur = (k0 >> 5) & 1;
    const int k1 = (k0 + 32) & (D_MODEL - 1);  // last iter wraps to 0 (dummy)
    // issue next B tile (other LDS buffer) + next A frags up front
    async_b128_to_lds(gB + k1, ldsBase + (unsigned)((cur ^ 1) * 4096) + ldsRow);
    v16bf na0 = load_frag_a(aRow0 + k1);
    v16bf na1 = load_frag_a(aRow1 + k1);
    // consume current fragments
#pragma unroll
    for (int t = 0; t < 4; ++t) {
      v16bf b = load_frag_b(
          (const __bf16*)((const char*)&Bs[cur][16 * t + (lane & 15)][0] + dB2));
      acc[0][t] = wmma_bf16(a0, b, acc[0][t]);
      acc[1][t] = wmma_bf16(a1, b, acc[1][t]);
    }
    a0 = na0;
    a1 = na1;
    // own async copies must land before signaling; DS reads already retired
    // (their consumers issued above). No vmem drain here.
    wait_asynccnt0();
    block_bar();
  }

  // C layout: VGPR r -> row (r + 8*hi), col = (lane&15) + 16*t
#pragma unroll
  for (int rb = 0; rb < 2; ++rb)
#pragma unroll
    for (int t = 0; t < 4; ++t)
#pragma unroll
      for (int r = 0; r < 8; ++r) {
        int row = m0 + rb * 16 + r + (hi ? 8 : 0);
        int col = n0 + 16 * t + (lane & 15);
        float v = acc[rb][t][r];
        if (mode == 0) {
          ((__bf16*)out)[(size_t)row * D_MODEL + col] = (__bf16)v;
        } else if (mode == 1) {
          int b_ = row >> 11, s_ = row & (SEQ - 1);
          int h_ = col >> 6, d_ = col & (DK - 1);
          ((__bf16*)out)[(((size_t)(b_ * NHEADS + h_) * DK + d_) * SEQ) + s_] =
              (__bf16)v;
        } else {
          ((float*)out)[(size_t)row * D_MODEL + col] = v;
        }
      }
}

// ---------------------------------------------------------------- attention
// One wave owns a 16-row q-tile of one (b,h); flash loop over key tiles of 32.
// K and V fragments are double-buffered in registers (loads lead by 1 iter).
__global__ __launch_bounds__(128) void flash_attn(
    const __bf16* __restrict__ Qp, const __bf16* __restrict__ Kp,
    const __bf16* __restrict__ Vt, __bf16* __restrict__ AO) {
  __shared__ __attribute__((aligned(64))) __bf16 Plds[4][16][32];

  const int lane = threadIdx.x & 31;
  const int wave = threadIdx.x >> 5;
  const int qblk = blockIdx.x & 31;          // S/64 = 32
  const int h    = (blockIdx.x >> 5) & 15;
  const int b    = blockIdx.x >> 9;
  const int q0   = qblk * 64 + wave * 16;
  const int hi   = (lane >= 16);
  const int dA   = hi ? 16 : 0;
  const int dB   = hi ? 16 : 0;
  const float scale = 0.125f;                // 1/sqrt(64)

  // Q A-fragments (16 x 64 -> two 16x32 frags), loaded once
  const __bf16* qRow =
      Qp + (size_t)(b * SEQ + q0 + (lane & 15)) * D_MODEL + h * DK;
  v16bf aq0 = load_frag_a(qRow + 0 + dA);
  v16bf aq1 = load_frag_a(qRow + 32 + dA);

  const __bf16* Kb  = Kp + (size_t)b * SEQ * D_MODEL + h * DK + dB;
  const __bf16* Vtb = Vt + (size_t)(b * NHEADS + h) * DK * SEQ + dB;
  const __bf16* kRowBase = Kb + (size_t)(lane & 15) * D_MODEL;
  const __bf16* vRowBase = Vtb + (size_t)(lane & 15) * SEQ;

  v8f acc[4];  // O accumulator, 16 x 64 f32
#pragma unroll
  for (int t = 0; t < 4; ++t) acc[t] = zero_v8f();
  float mrow[8], lrow[8];
#pragma unroll
  for (int r = 0; r < 8; ++r) { mrow[r] = -1.0e30f; lrow[r] = 0.0f; }

  // prologue: fragments for s0 = 0
  v16bf kb[4], vb[4];
#pragma unroll
  for (int t = 0; t < 2; ++t) {
    const __bf16* kR = kRowBase + (size_t)(16 * t) * D_MODEL;
    kb[2 * t + 0] = load_frag_b(kR);
    kb[2 * t + 1] = load_frag_b(kR + 32);
  }
#pragma unroll
  for (int t = 0; t < 4; ++t)
    vb[t] = load_frag_b(vRowBase + (size_t)(16 * t) * SEQ);

  for (int s0 = 0; s0 < SEQ; s0 += 32) {
    const int s1 = (s0 + 32) & (SEQ - 1);  // last iter wraps to 0 (dummy)
    // issue next-iteration K/V fragment loads up front
    v16bf nkb[4], nvb[4];
#pragma unroll
    for (int t = 0; t < 2; ++t) {
      const __bf16* kR = kRowBase + (size_t)(s1 + 16 * t) * D_MODEL;
      nkb[2 * t + 0] = load_frag_b(kR);
      nkb[2 * t + 1] = load_frag_b(kR + 32);
    }
#pragma unroll
    for (int t = 0; t < 4; ++t)
      nvb[t] = load_frag_b(vRowBase + (size_t)(16 * t) * SEQ + s1);

    // ---- scores: two 16x16 C tiles over 32 keys, contract d = 64
    v8f st[2];
#pragma unroll
    for (int t = 0; t < 2; ++t) {
      st[t] = zero_v8f();
      st[t] = wmma_bf16(aq0, kb[2 * t + 0], st[t]);
      st[t] = wmma_bf16(aq1, kb[2 * t + 1], st[t]);
    }

    // ---- streaming softmax (per row r; half-wave shuffle reductions)
#pragma unroll
    for (int r = 0; r < 8; ++r) {
      float s0v = st[0][r] * scale;
      float s1v = st[1][r] * scale;
      float mx = fmaxf(s0v, s1v);
      mx = fmaxf(mx, __shfl_xor(mx, 1));
      mx = fmaxf(mx, __shfl_xor(mx, 2));
      mx = fmaxf(mx, __shfl_xor(mx, 4));
      mx = fmaxf(mx, __shfl_xor(mx, 8));
      float mnew = fmaxf(mrow[r], mx);
      float corr = __expf(mrow[r] - mnew);
      float p0 = __expf(s0v - mnew);
      float p1 = __expf(s1v - mnew);
      float rs = p0 + p1;
      rs += __shfl_xor(rs, 1);
      rs += __shfl_xor(rs, 2);
      rs += __shfl_xor(rs, 4);
      rs += __shfl_xor(rs, 8);
      lrow[r] = lrow[r] * corr + rs;
      mrow[r] = mnew;
      st[0][r] = p0;
      st[1][r] = p1;
#pragma unroll
      for (int t = 0; t < 4; ++t) acc[t][r] *= corr;
    }

    // ---- P (C-layout) -> wave-private LDS slice -> A-layout fragment.
    // Wave-internal DS ops are in-order; only a scheduling fence is needed.
    wavebar();
#pragma unroll
    for (int t = 0; t < 2; ++t)
#pragma unroll
      for (int r = 0; r < 8; ++r)
        Plds[wave][r + (hi ? 8 : 0)][(lane & 15) + 16 * t] = (__bf16)st[t][r];
    wavebar();
    v16bf ap = load_frag_a(&Plds[wave][lane & 15][dA]);

    // ---- O += P * V  (contract 32 keys; Vt gives contiguous B-frags)
#pragma unroll
    for (int t = 0; t < 4; ++t) acc[t] = wmma_bf16(ap, vb[t], acc[t]);

#pragma unroll
    for (int t = 0; t < 4; ++t) { kb[t] = nkb[t]; vb[t] = nvb[t]; }
  }

  // ---- normalize and store merged heads (bf16)
#pragma unroll
  for (int r = 0; r < 8; ++r) {
    float inv = 1.0f / lrow[r];
#pragma unroll
    for (int t = 0; t < 4; ++t) acc[t][r] *= inv;
  }
#pragma unroll
  for (int t = 0; t < 4; ++t)
#pragma unroll
    for (int r = 0; r < 8; ++r) {
      int row = q0 + r + (hi ? 8 : 0);
      int col = h * DK + 16 * t + (lane & 15);
      AO[(size_t)(b * SEQ + row) * D_MODEL + col] = (__bf16)acc[t][r];
    }
}

// ---------------------------------------------------------------- launcher
extern "C" void kernel_launch(void* const* d_in, const int* in_sizes, int n_in,
                              void* d_out, int out_size, void* d_ws,
                              size_t ws_size, hipStream_t stream) {
  const float* q  = (const float*)d_in[0];
  const float* k  = (const float*)d_in[1];
  const float* v  = (const float*)d_in[2];
  const float* Wq = (const float*)d_in[3];
  const float* Wk = (const float*)d_in[4];
  const float* Wv = (const float*)d_in[5];
  const float* Wo = (const float*)d_in[6];
  float* out = (float*)d_out;

  const size_t MB = 1u << 20;
  char* ws = (char*)d_ws;
  __bf16* qb  = (__bf16*)(ws + 0 * MB);
  __bf16* kb  = (__bf16*)(ws + 8 * MB);
  __bf16* vb  = (__bf16*)(ws + 16 * MB);
  __bf16* wqb = (__bf16*)(ws + 24 * MB);
  __bf16* wkb = (__bf16*)(ws + 26 * MB);
  __bf16* wvb = (__bf16*)(ws + 28 * MB);
  __bf16* wob = (__bf16*)(ws + 30 * MB);
  __bf16* Qp  = (__bf16*)(ws + 32 * MB);
  __bf16* Kp  = (__bf16*)(ws + 40 * MB);
  __bf16* Vt  = (__bf16*)(ws + 48 * MB);
  __bf16* AO  = (__bf16*)(ws + 56 * MB);

  const int ACT = BATCH * SEQ * D_MODEL;   // 4 Mi elements
  const int WEL = D_MODEL * D_MODEL;       // 1 Mi elements

  cvt_f32_bf16<<<2048, 256, 0, stream>>>(q, qb, ACT);
  cvt_f32_bf16<<<2048, 256, 0, stream>>>(k, kb, ACT);
  cvt_f32_bf16<<<2048, 256, 0, stream>>>(v, vb, ACT);
  cvt_f32_bf16<<<1024, 256, 0, stream>>>(Wq, wqb, WEL);
  cvt_f32_bf16<<<1024, 256, 0, stream>>>(Wk, wkb, WEL);
  cvt_f32_bf16<<<1024, 256, 0, stream>>>(Wv, wvb, WEL);
  cvt_f32_bf16<<<1024, 256, 0, stream>>>(Wo, wob, WEL);

  dim3 gGemm(BATCH * SEQ / 256, D_MODEL / 64);  // (16, 16)
  gemm_wmma<<<gGemm, 256, 0, stream>>>(qb, wqb, Qp, 0);
  gemm_wmma<<<gGemm, 256, 0, stream>>>(kb, wkb, Kp, 0);
  gemm_wmma<<<gGemm, 256, 0, stream>>>(vb, wvb, Vt, 1);

  flash_attn<<<BATCH * NHEADS * (SEQ / 64), 128, 0, stream>>>(Qp, Kp, Vt, AO);

  gemm_wmma<<<gGemm, 256, 0, stream>>>(AO, wob, out, 2);
}
